// SpatioTemporalNet_87857851007245
// MI455X (gfx1250) — compile-verified, matrix operand order checked by reading, not run on hardware
//
#include <hip/hip_runtime.h>

typedef __attribute__((ext_vector_type(16))) _Float16 v16h;
typedef __attribute__((ext_vector_type(8)))  _Float16 v8h;
typedef __attribute__((ext_vector_type(8)))  float    v8f;

#define NVOX 4500
#define BT   80
#define TSTEPS 5
#define CDIV(a,b) (((a)+(b)-1)/(b))

__device__ __forceinline__ float stn_sigmoid(float x) { return 1.f / (1.f + expf(-x)); }

// ---------------------------------------------------------------------------
// Pack weights into per-lane WMMA B-fragment order.
// B[k][n] = W[n*CIN + k] (GEMM, TAPS==1) or W[(n*CIN + k)*TAPS + tap] (conv).
// Fragment layout (16-bit, 16x16x32): lane l holds column n = l%16; K base =
// kc*32 + (l<16 ? 0 : 8); elements 0..7 -> K=kb..kb+7, 8..15 -> K=kb+16..kb+23.
// pack[((nt*TAPS + tap)*KC + kc)*32 + lane][16]
// ---------------------------------------------------------------------------
__global__ void stn_pack_weights(const float* __restrict__ W, _Float16* __restrict__ pack,
                                 int COUT, int CIN, int TAPS, int CINP) {
  int KC = CINP / 32;
  int NT = COUT / 16;
  int total = NT * TAPS * KC * 32;
  int idx = blockIdx.x * blockDim.x + threadIdx.x;
  if (idx >= total) return;
  int lane = idx & 31;
  int r = idx >> 5;
  int kc = r % KC; r /= KC;
  int tap = r % TAPS; r /= TAPS;
  int nt = r;
  int n = nt * 16 + (lane & 15);
  int kbase = kc * 32 + ((lane < 16) ? 0 : 8);
  _Float16* dst = pack + ((((size_t)nt * TAPS + tap) * KC + kc) * 32 + lane) * 16;
  for (int e = 0; e < 16; ++e) {
    int k = kbase + ((e < 8) ? e : (16 + e - 8));
    float val = (k < CIN) ? W[((size_t)n * CIN + k) * TAPS + tap] : 0.f;
    dst[e] = (_Float16)val;
  }
}

// x (80,2,4500) f32 -> xf16 (80,32,4500) f16, channels >=2 zero-padded
__global__ void stn_convert_x(const float* __restrict__ x, _Float16* __restrict__ xf) {
  size_t idx = (size_t)blockIdx.x * blockDim.x + threadIdx.x;
  size_t total = (size_t)BT * 32 * NVOX;
  if (idx >= total) return;
  int v = (int)(idx % NVOX);
  size_t r = idx / NVOX;
  int c = (int)(r % 32);
  int f = (int)(r / 32);
  float val = (c < 2) ? x[((size_t)f * 2 + c) * NVOX + v] : 0.f;
  xf[idx] = (_Float16)val;
}

// ---------------------------------------------------------------------------
// Implicit-GEMM 3D conv (3x3x3 pad=1 when TAPS==27, 1x1x1 when TAPS==1),
// fully unrolled over taps/K-chunks. One wave computes a 16-voxel x COUT
// stripe: the gathered A-fragment is reused across all NT n-tiles (NT WMMAs
// per gather). f16 inputs, f32 accumulation.
// ---------------------------------------------------------------------------
template <int TAPS, int KC, int NT>
__global__ void stn_conv_wmma_t(const _Float16* __restrict__ act, const _Float16* __restrict__ pack,
                                const float* __restrict__ bias, float* __restrict__ out) {
  constexpr int CINP = KC * 32;
  constexpr int COUT = NT * 16;
  const int lane = threadIdx.x;
  const int mt = blockIdx.x, f = blockIdx.y;
  const _Float16* A = act + (size_t)f * CINP * NVOX;
  int v = mt * 16 + (lane & 15);
  if (v >= NVOX) v = NVOX - 1;
  int w = v % 5, h = (v / 5) % 15, d = v / 75;
  int koff = (lane < 16) ? 0 : 8;
  v8f acc[NT];
#pragma unroll
  for (int i = 0; i < NT; ++i) acc[i] = (v8f){};
#pragma unroll
  for (int tap = 0; tap < TAPS; ++tap) {
    const int kd = (TAPS == 27) ? (tap / 9 - 1) : 0;
    const int kh = (TAPS == 27) ? ((tap / 3) % 3 - 1) : 0;
    const int kw = (TAPS == 27) ? (tap % 3 - 1) : 0;
    int nd = d + kd, nh = h + kh, nw = w + kw;
    bool inb = (nd >= 0 && nd < 60 && nh >= 0 && nh < 15 && nw >= 0 && nw < 5);
    int nv = inb ? ((nd * 15 + nh) * 5 + nw) : 0;
#pragma unroll
    for (int kc = 0; kc < KC; ++kc) {
      int kb = kc * 32 + koff;
      const _Float16* Ap = A + (size_t)kb * NVOX + nv;
      v16h a;
#pragma unroll
      for (int j = 0; j < 8; ++j) a[j] = inb ? Ap[(size_t)j * NVOX] : (_Float16)0.f;
#pragma unroll
      for (int j = 0; j < 8; ++j) a[8 + j] = inb ? Ap[(size_t)(16 + j) * NVOX] : (_Float16)0.f;
#pragma unroll
      for (int nt = 0; nt < NT; ++nt) {
        const _Float16* bp = pack + ((((size_t)nt * TAPS + tap) * KC + kc) * 32 + lane) * 16;
        __builtin_prefetch(bp + 512 * 16, 0, 0);
        v16h b = *(const v16h*)bp;
        acc[nt] = __builtin_amdgcn_wmma_f32_16x16x32_f16(false, a, false, b, (short)0, acc[nt], false, false);
      }
    }
  }
  float* Of = out + (size_t)f * COUT * NVOX;
#pragma unroll
  for (int nt = 0; nt < NT; ++nt) {
    int n = nt * 16 + (lane & 15);
    float bv = bias[n];
    float* O = Of + (size_t)n * NVOX;
#pragma unroll
    for (int g = 0; g < 8; ++g) {
      int vv = mt * 16 + g + ((lane < 16) ? 0 : 8);
      if (vv < NVOX) O[vv] = acc[nt][g] + bv;
    }
  }
}

// Per-channel mean/var over (F, SP) of raw[f][c][SP]
__global__ void stn_bn_stats(const float* __restrict__ raw, float* __restrict__ stats,
                             int C, int SP, int F) {
  int c = blockIdx.x;
  float s = 0.f, ss = 0.f;
  long n = (long)F * SP;
  for (long i = threadIdx.x; i < n; i += blockDim.x) {
    int f = (int)(i / SP);
    int v = (int)(i % SP);
    float x = raw[((size_t)f * C + c) * SP + v];
    s += x; ss += x * x;
  }
  __shared__ float sh[256];
  __shared__ float sh2[256];
  sh[threadIdx.x] = s; sh2[threadIdx.x] = ss;
  __syncthreads();
  for (int st = 128; st > 0; st >>= 1) {
    if ((int)threadIdx.x < st) { sh[threadIdx.x] += sh[threadIdx.x + st]; sh2[threadIdx.x] += sh2[threadIdx.x + st]; }
    __syncthreads();
  }
  if (threadIdx.x == 0) {
    float nf = (float)n;
    float m = sh[0] / nf;
    float var = sh2[0] / nf - m * m;
    stats[2 * c] = m; stats[2 * c + 1] = var;
  }
}

// raw f32 [F][C][SP] -> BN+ReLU -> out f16 [F][CP][SP] (channels >= C zeroed)
__global__ void stn_bn_apply_f16(const float* __restrict__ raw, const float* __restrict__ stats,
                                 const float* __restrict__ g, const float* __restrict__ bb,
                                 _Float16* __restrict__ out, int C, int CP, int SP, int F) {
  size_t idx = (size_t)blockIdx.x * blockDim.x + threadIdx.x;
  size_t total = (size_t)F * CP * SP;
  if (idx >= total) return;
  int v = (int)(idx % SP);
  size_t r = idx / SP;
  int c = (int)(r % CP);
  int f = (int)(r / CP);
  float val = 0.f;
  if (c < C) {
    float x = raw[((size_t)f * C + c) * SP + v];
    float m = stats[2 * c], var = stats[2 * c + 1];
    val = (x - m) * rsqrtf(var + 1e-5f) * g[c] + bb[c];
    val = fmaxf(val, 0.f);
  }
  out[idx] = (_Float16)val;
}

// in-place BN + ReLU for decoder stages (f32)
__global__ void stn_bn_apply_f32(float* __restrict__ buf, const float* __restrict__ stats,
                                 const float* __restrict__ g, const float* __restrict__ bb,
                                 int C, int SP, int F) {
  size_t idx = (size_t)blockIdx.x * blockDim.x + threadIdx.x;
  size_t total = (size_t)F * C * SP;
  if (idx >= total) return;
  size_t r = idx / SP;
  int c = (int)(r % C);
  float x = buf[idx];
  float m = stats[2 * c], var = stats[2 * c + 1];
  float val = (x - m) * rsqrtf(var + 1e-5f) * g[c] + bb[c];
  buf[idx] = fmaxf(val, 0.f);
}

// avg-pool einsum: (80,32,60,15,5) -> feats (80,2048) f32 + f16 mirror
__global__ void stn_pool(const _Float16* __restrict__ act, float* __restrict__ feats,
                         _Float16* __restrict__ featsh) {
  int idx = blockIdx.x * blockDim.x + threadIdx.x;
  if (idx >= BT * 2048) return;
  int rr = idx % 2;
  int q = (idx / 2) % 4;
  int p = (idx / 8) % 8;
  int c = (idx / 64) % 32;
  int f = idx / 2048;
  int sD = p * 60 / 8, eD = ((p + 1) * 60 + 7) / 8;
  int sH = q * 15 / 4, eH = ((q + 1) * 15 + 3) / 4;
  int sW = rr * 5 / 2, eW = ((rr + 1) * 5 + 1) / 2;
  float scale = 1.f / ((float)(eD - sD) * (eH - sH) * (eW - sW));
  const _Float16* a = act + ((size_t)f * 32 + c) * NVOX;
  float s = 0.f;
  for (int d = sD; d < eD; ++d)
    for (int h = sH; h < eH; ++h)
      for (int w = sW; w < eW; ++w)
        s += (float)a[(d * 15 + h) * 5 + w];
  float o = s * scale;
  feats[idx] = o;
  featsh[idx] = (_Float16)o;
}

// ---------------------------------------------------------------------------
// C[M,N] f32 = A(f16, M x K row-major, lda) x packed-B.
// One wave per 16-row x (NTB*16)-col stripe: A fragment (two 16B vector
// loads, K contiguous) is reused across NTB WMMAs.
// ---------------------------------------------------------------------------
template <int NTB>
__global__ void stn_gemm_wmma_t(const _Float16* __restrict__ A, const _Float16* __restrict__ pack,
                                float* __restrict__ out, int M, int N, int K, int lda, int ldc) {
  const int lane = threadIdx.x;
  const int mt = blockIdx.x;
  const int ntb = blockIdx.y;
  const int KC = K >> 5;
  int mr = mt * 16 + (lane & 15);
  if (mr >= M) mr = M - 1;
  const _Float16* Ar = A + (size_t)mr * lda;
  int koff = (lane < 16) ? 0 : 8;
  v8f acc[NTB];
#pragma unroll
  for (int i = 0; i < NTB; ++i) acc[i] = (v8f){};
#pragma unroll 4
  for (int kc = 0; kc < KC; ++kc) {
    int kb = kc * 32 + koff;
    v8h lo = *(const v8h*)(Ar + kb);
    v8h hi = *(const v8h*)(Ar + kb + 16);
    v16h a = __builtin_shufflevector(lo, hi, 0, 1, 2, 3, 4, 5, 6, 7, 8, 9, 10, 11, 12, 13, 14, 15);
#pragma unroll
    for (int q = 0; q < NTB; ++q) {
      int nt = ntb * NTB + q;
      const _Float16* bp = pack + (((size_t)nt * KC + kc) * 32 + lane) * 16;
      __builtin_prefetch(bp + 512 * 16, 0, 0);
      v16h b = *(const v16h*)bp;
      acc[q] = __builtin_amdgcn_wmma_f32_16x16x32_f16(false, a, false, b, (short)0, acc[q], false, false);
    }
  }
#pragma unroll
  for (int q = 0; q < NTB; ++q) {
    int n = (ntb * NTB + q) * 16 + (lane & 15);
#pragma unroll
    for (int g = 0; g < 8; ++g) {
      int r = mt * 16 + g + ((lane < 16) ? 0 : 8);
      if (r < M && n < N) out[(size_t)r * ldc + n] = acc[q][g];
    }
  }
}

// LSTM gate fusion for one time step (16 batch x 128 hidden)
__global__ void stn_lstm_gate(const float* __restrict__ gi, int giStride,
                              const float* __restrict__ gh,
                              const float* __restrict__ bih, const float* __restrict__ bhh,
                              float* __restrict__ cstate, _Float16* __restrict__ hstate,
                              _Float16* __restrict__ hseq, int t) {
  int idx = blockIdx.x * blockDim.x + threadIdx.x;
  if (idx >= 16 * 128) return;
  int b = idx >> 7, j = idx & 127;
  const float* gr = gi + (size_t)b * giStride;
  const float* ghr = gh + (size_t)b * 512;
  float vi = gr[j]       + ghr[j]       + bih[j]       + bhh[j];
  float vf = gr[128 + j] + ghr[128 + j] + bih[128 + j] + bhh[128 + j];
  float vg = gr[256 + j] + ghr[256 + j] + bih[256 + j] + bhh[256 + j];
  float vo = gr[384 + j] + ghr[384 + j] + bih[384 + j] + bhh[384 + j];
  float cp = (t == 0) ? 0.f : cstate[idx];
  float c = stn_sigmoid(vf) * cp + stn_sigmoid(vi) * tanhf(vg);
  float h = stn_sigmoid(vo) * tanhf(c);
  cstate[idx] = c;
  hstate[idx] = (_Float16)h;
  if (hseq) hseq[((size_t)b * TSTEPS + t) * 128 + j] = (_Float16)h;
}

__global__ void stn_add_bias(float* __restrict__ y, const float* __restrict__ b) {
  int idx = blockIdx.x * blockDim.x + threadIdx.x;
  if (idx >= BT * 2048) return;
  y[idx] += b[idx % 2048];
}

// Transposed conv, stride 2, kernel 4, pad 1 (gather form).
// W layout is the reference's (CI, CO, 4,4,4); uses flipped taps.
__global__ void stn_convT(const float* __restrict__ in, const float* __restrict__ W,
                          const float* __restrict__ bias, float* __restrict__ out,
                          int F, int CI, int CO, int ID, int IH, int IW, int sigmoidOut) {
  int OD = 2 * ID, OH = 2 * IH, OW = 2 * IW;
  size_t total = (size_t)F * CO * OD * OH * OW;
  size_t idx = (size_t)blockIdx.x * blockDim.x + threadIdx.x;
  if (idx >= total) return;
  int ow = (int)(idx % OW); size_t r = idx / OW;
  int oh = (int)(r % OH); r /= OH;
  int od = (int)(r % OD); r /= OD;
  int co = (int)(r % CO);
  int f = (int)(r / CO);
  int kd[2], di[2], nd = 0;
  int kh[2], hi[2], nh = 0;
  int kw[2], wi[2], nw = 0;
  for (int k = od & 1; k < 4; k += 2) { int p = od + k - 2; if (p >= 0) { int i = p >> 1; if (i < ID) { kd[nd] = k; di[nd] = i; nd++; } } }
  for (int k = oh & 1; k < 4; k += 2) { int p = oh + k - 2; if (p >= 0) { int i = p >> 1; if (i < IH) { kh[nh] = k; hi[nh] = i; nh++; } } }
  for (int k = ow & 1; k < 4; k += 2) { int p = ow + k - 2; if (p >= 0) { int i = p >> 1; if (i < IW) { kw[nw] = k; wi[nw] = i; nw++; } } }
  float acc = bias[co];
  const float* inF = in + (size_t)f * CI * ID * IH * IW;
  for (int ci = 0; ci < CI; ++ci) {
    const float* ip = inF + (size_t)ci * ID * IH * IW;
    const float* wp = W + ((size_t)ci * CO + co) * 64;
    for (int a = 0; a < nd; ++a)
      for (int b2 = 0; b2 < nh; ++b2)
        for (int c2 = 0; c2 < nw; ++c2)
          acc += ip[(di[a] * IH + hi[b2]) * IW + wi[c2]] *
                 wp[(3 - kd[a]) * 16 + (3 - kh[b2]) * 4 + (3 - kw[c2])];
  }
  out[idx] = sigmoidOut ? stn_sigmoid(acc) : acc;
}

// trilinear interp einsum (64,32,16) -> (60,15,5); only t<3 frames are produced
__global__ void stn_interp(const float* __restrict__ d3, float* __restrict__ frames) {
  int idx = blockIdx.x * blockDim.x + threadIdx.x;
  if (idx >= 48 * 2 * NVOX) return;
  int v = idx % NVOX;
  int c = (idx / NVOX) & 1;
  int fo = idx / (2 * NVOX);
  int b = fo / 3, t = fo % 3;
  int fsrc = b * 5 + t;
  int w = v % 5, h = (v / 5) % 15, d = v / 75;
  float sd = d * (63.f / 59.f); int dlo = (int)floorf(sd); if (dlo > 63) dlo = 63; float wd = sd - dlo; int dhi = dlo + 1; if (dhi > 63) dhi = 63;
  float shh = h * (31.f / 14.f); int hlo = (int)floorf(shh); if (hlo > 31) hlo = 31; float wh = shh - hlo; int hhi = hlo + 1; if (hhi > 31) hhi = 31;
  float sw = w * (15.f / 4.f);  int wlo = (int)floorf(sw); if (wlo > 15) wlo = 15; float ww = sw - wlo; int whi = wlo + 1; if (whi > 15) whi = 15;
  const float* src = d3 + ((size_t)fsrc * 2 + c) * 32768;
  float a000 = src[(dlo * 32 + hlo) * 16 + wlo];
  float a001 = src[(dlo * 32 + hlo) * 16 + whi];
  float a010 = src[(dlo * 32 + hhi) * 16 + wlo];
  float a011 = src[(dlo * 32 + hhi) * 16 + whi];
  float a100 = src[(dhi * 32 + hlo) * 16 + wlo];
  float a101 = src[(dhi * 32 + hlo) * 16 + whi];
  float a110 = src[(dhi * 32 + hhi) * 16 + wlo];
  float a111 = src[(dhi * 32 + hhi) * 16 + whi];
  float acc = (1.f - wd) * ((1.f - wh) * ((1.f - ww) * a000 + ww * a001) + wh * ((1.f - ww) * a010 + ww * a011))
            +        wd  * ((1.f - wh) * ((1.f - ww) * a100 + ww * a101) + wh * ((1.f - ww) * a110 + ww * a111));
  frames[idx] = acc;
}

// bitonic argsort (descending energy, stable) per frame; LDS 64KB of u64 keys
__global__ void stn_sort(const float* __restrict__ frames, int* __restrict__ orderG) {
  __shared__ unsigned long long sh[8192];
  int fo = blockIdx.x;
  int tid = threadIdx.x;
  const float* e = frames + (size_t)fo * 2 * NVOX;  // channel 0
  for (int i = tid; i < 8192; i += 1024) {
    unsigned long long comp;
    if (i < NVOX) {
      unsigned u = __float_as_uint(e[i]);
      u = (u & 0x80000000u) ? ~u : (u | 0x80000000u);
      comp = ((unsigned long long)(~u) << 32) | (unsigned)i;  // ascending ~u == descending energy
    } else {
      comp = 0xFFFFFFFFFFFFFFFFull;
    }
    sh[i] = comp;
  }
  __syncthreads();
  for (int k = 2; k <= 8192; k <<= 1) {
    for (int j = k >> 1; j > 0; j >>= 1) {
      for (int i = tid; i < 8192; i += 1024) {
        int l = i ^ j;
        if (l > i) {
          bool up = ((i & k) == 0);
          unsigned long long a = sh[i], b = sh[l];
          if ((a > b) == up) { sh[i] = b; sh[l] = a; }
        }
      }
      __syncthreads();
    }
  }
  for (int i = tid; i < NVOX; i += 1024)
    orderG[fo * NVOX + i] = (int)(sh[i] & 0xFFFFFFFFu);
}

// Greedy NMS: dist^2 < 4 on integer grid == 27-neighborhood suppression.
// One wave per frame: sequential over sorted order, lanes check the 27 offsets.
__global__ void stn_nms(const int* __restrict__ orderG, unsigned char* __restrict__ maskG) {
  __shared__ unsigned char kept[NVOX];
  __shared__ unsigned char posKeep[NVOX];
  __shared__ int shcount;
  int fo = blockIdx.x;
  int lane = threadIdx.x;
  for (int i = lane; i < NVOX; i += 32) { kept[i] = 0; posKeep[i] = 0; }
  if (lane == 0) shcount = 0;
  __syncthreads();
  const int* ord = orderG + fo * NVOX;
  for (int i = 0; i < NVOX; ++i) {
    int v = ord[i];
    int w = v % 5, h = (v / 5) % 15, d = v / 75;
    int sup = 0;
    if (lane < 27) {
      int od = lane / 9 - 1, oh = (lane / 3) % 3 - 1, ow = lane % 3 - 1;
      int nd = d + od, nh = h + oh, nw = w + ow;
      if (nd >= 0 && nd < 60 && nh >= 0 && nh < 15 && nw >= 0 && nw < 5)
        sup = kept[(nd * 15 + nh) * 5 + nw];
    }
    bool any = (__ballot(sup) != 0ull);
    if (lane == 0) {
      unsigned char kp = any ? 0 : 1;
      kept[v] = kp;
      posKeep[i] = kp;
      shcount += kp;
    }
    __syncthreads();
  }
  int cnt = shcount;
  for (int i = lane; i < NVOX; i += 32) {
    unsigned char kp = posKeep[i];
    if (cnt > 100 && i >= 100) kp = 0;
    maskG[fo * NVOX + ord[i]] = kp;
  }
}

__global__ void stn_final(const float* __restrict__ frames, const unsigned char* __restrict__ maskG,
                          float* __restrict__ out) {
  int idx = blockIdx.x * blockDim.x + threadIdx.x;
  if (idx >= 48 * 2 * NVOX) return;
  int v = idx % NVOX;
  int fo = idx / (2 * NVOX);
  out[idx] = frames[idx] * (maskG[fo * NVOX + v] ? 1.f : 0.f);
}

// ---------------------------------------------------------------------------
extern "C" void kernel_launch(void* const* d_in, const int* in_sizes, int n_in,
                              void* d_out, int out_size, void* d_ws, size_t ws_size,
                              hipStream_t stream) {
  (void)in_sizes; (void)n_in; (void)out_size; (void)ws_size;

  const float* X      = (const float*)d_in[0];
  const float* e1w    = (const float*)d_in[1];
  const float* e1b    = (const float*)d_in[2];
  const float* e1g    = (const float*)d_in[3];
  const float* e1bb   = (const float*)d_in[4];
  const float* e2w    = (const float*)d_in[5];
  const float* e2b    = (const float*)d_in[6];
  const float* e2g    = (const float*)d_in[7];
  const float* e2bb   = (const float*)d_in[8];
  const float* e3w    = (const float*)d_in[9];
  const float* e3b    = (const float*)d_in[10];
  const float* e3g    = (const float*)d_in[11];
  const float* e3bb   = (const float*)d_in[12];
  const float* c1w    = (const float*)d_in[13];
  const float* c1b    = (const float*)d_in[14];
  const float* c1g    = (const float*)d_in[15];
  const float* c1bb   = (const float*)d_in[16];
  const float* c2w    = (const float*)d_in[17];
  const float* c2b    = (const float*)d_in[18];
  const float* c2g    = (const float*)d_in[19];
  const float* c2bb   = (const float*)d_in[20];
  const float* l0_wih = (const float*)d_in[21];
  const float* l0_whh = (const float*)d_in[22];
  const float* l0_bih = (const float*)d_in[23];
  const float* l0_bhh = (const float*)d_in[24];
  const float* l1_wih = (const float*)d_in[25];
  const float* l1_whh = (const float*)d_in[26];
  const float* l1_bih = (const float*)d_in[27];
  const float* l1_bhh = (const float*)d_in[28];
  const float* fcw    = (const float*)d_in[29];
  const float* fcb    = (const float*)d_in[30];
  const float* d1w    = (const float*)d_in[31];
  const float* d1b    = (const float*)d_in[32];
  const float* d1g    = (const float*)d_in[33];
  const float* d1bb   = (const float*)d_in[34];
  const float* d2w    = (const float*)d_in[35];
  const float* d2b    = (const float*)d_in[36];
  const float* d2g    = (const float*)d_in[37];
  const float* d2bb   = (const float*)d_in[38];
  const float* d3w    = (const float*)d_in[39];
  const float* d3b    = (const float*)d_in[40];

  // ---- workspace layout (region aliasing by lifetime) ----
  char* base = (char*)d_ws;
  size_t off = 0;
  auto alloc = [&](size_t bytes) -> char* {
    off = (off + 255) & ~(size_t)255;
    char* p = base + off; off += bytes; return p;
  };
  char* R0 = alloc(46080000);  // xf16 | a2raw | c1raw
  char* R1 = alloc(23040000);  // a1raw | a2f16 | c1f16
  char* R2 = alloc(23040000);  // a1f16 | c2f16
  char* R3 = alloc(92160000);  // a3raw | decoder bufs + frames
  char* R4 = alloc(46080000);  // a3f16 | c2raw
  float*    stats  = (float*)alloc(128 * 2 * sizeof(float));
  float*    feats  = (float*)alloc((size_t)BT * 2048 * sizeof(float));
  _Float16* featsh = (_Float16*)alloc((size_t)BT * 2048 * sizeof(_Float16));
  float*    gi     = (float*)alloc((size_t)BT * 512 * sizeof(float));
  float*    gh1    = (float*)alloc(16 * 512 * sizeof(float));
  float*    gi2    = (float*)alloc(16 * 512 * sizeof(float));
  float*    gh2    = (float*)alloc(16 * 512 * sizeof(float));
  _Float16* h1st   = (_Float16*)alloc(16 * 128 * sizeof(_Float16));
  _Float16* h2st   = (_Float16*)alloc(16 * 128 * sizeof(_Float16));
  float*    c1st   = (float*)alloc(16 * 128 * sizeof(float));
  float*    c2st   = (float*)alloc(16 * 128 * sizeof(float));
  _Float16* h2seq  = (_Float16*)alloc((size_t)BT * 128 * sizeof(_Float16));
  float*    yb     = (float*)alloc((size_t)BT * 2048 * sizeof(float));
  int*      orderG = (int*)alloc((size_t)48 * NVOX * sizeof(int));
  unsigned char* maskG = (unsigned char*)alloc((size_t)48 * NVOX);
  _Float16* pe1  = (_Float16*)alloc((size_t)1 * 27 * 1 * 512 * 2);
  _Float16* pe2  = (_Float16*)alloc((size_t)2 * 27 * 1 * 512 * 2);
  _Float16* pe3  = (_Float16*)alloc((size_t)4 * 27 * 1 * 512 * 2);
  _Float16* pc1  = (_Float16*)alloc((size_t)2 * 1 * 2 * 512 * 2);
  _Float16* pc2  = (_Float16*)alloc((size_t)2 * 1 * 1 * 512 * 2);
  _Float16* pl0i = (_Float16*)alloc((size_t)32 * 64 * 512 * 2);
  _Float16* pl0h = (_Float16*)alloc((size_t)32 * 4 * 512 * 2);
  _Float16* pl1i = (_Float16*)alloc((size_t)32 * 4 * 512 * 2);
  _Float16* pl1h = (_Float16*)alloc((size_t)32 * 4 * 512 * 2);
  _Float16* pfc  = (_Float16*)alloc((size_t)128 * 4 * 512 * 2);

  _Float16* xf16  = (_Float16*)R0;
  float*    a1raw = (float*)R1;
  _Float16* a1f16 = (_Float16*)R2;
  float*    a2raw = (float*)R0;
  _Float16* a2f16 = (_Float16*)R1;
  float*    a3raw = (float*)R3;
  _Float16* a3f16 = (_Float16*)R4;
  float*    c1raw = (float*)R0;
  _Float16* c1f16 = (_Float16*)R1;
  float*    c2raw = (float*)R4;
  _Float16* c2f16 = (_Float16*)R2;
  float*    d1buf = (float*)R3;
  float*    d2buf = (float*)(R3 + 5242880);
  float*    d3buf = (float*)(R3 + 5242880 + 20971520);
  float*    frames = (float*)(R3 + 5242880 + 20971520 + 20971520);

  auto packLaunch = [&](const float* W, _Float16* pk, int COUT, int CIN, int TAPS, int CINP) {
    int total = (COUT / 16) * TAPS * (CINP / 32) * 32;
    stn_pack_weights<<<dim3(CDIV(total, 256)), dim3(256), 0, stream>>>(W, pk, COUT, CIN, TAPS, CINP);
  };
  auto bnStage = [&](const float* raw, const float* g, const float* bbias, _Float16* outf, int C, int CP) {
    stn_bn_stats<<<dim3(C), dim3(256), 0, stream>>>(raw, stats, C, NVOX, BT);
    size_t tot = (size_t)BT * CP * NVOX;
    stn_bn_apply_f16<<<dim3((unsigned)CDIV(tot, 256)), dim3(256), 0, stream>>>(raw, stats, g, bbias, outf, C, CP, NVOX, BT);
  };

  // weight packing (deterministic, re-run each call)
  packLaunch(e1w, pe1, 16, 2, 27, 32);
  packLaunch(e2w, pe2, 32, 16, 27, 32);
  packLaunch(e3w, pe3, 64, 32, 27, 32);
  packLaunch(c1w, pc1, 32, 64, 1, 64);
  packLaunch(c2w, pc2, 32, 32, 1, 32);
  packLaunch(l0_wih, pl0i, 512, 2048, 1, 2048);
  packLaunch(l0_whh, pl0h, 512, 128, 1, 128);
  packLaunch(l1_wih, pl1i, 512, 128, 1, 128);
  packLaunch(l1_whh, pl1h, 512, 128, 1, 128);
  packLaunch(fcw, pfc, 2048, 128, 1, 128);

  {
    size_t tot = (size_t)BT * 32 * NVOX;
    stn_convert_x<<<dim3((unsigned)CDIV(tot, 256)), dim3(256), 0, stream>>>(X, xf16);
  }

  // encoder (templated WMMA conv: one wave -> 16 voxels x all couts)
  dim3 cgrid(CDIV(NVOX, 16), BT);
  stn_conv_wmma_t<27, 1, 1><<<cgrid, dim3(32), 0, stream>>>(xf16, pe1, e1b, a1raw);
  bnStage(a1raw, e1g, e1bb, a1f16, 16, 32);
  stn_conv_wmma_t<27, 1, 2><<<cgrid, dim3(32), 0, stream>>>(a1f16, pe2, e2b, a2raw);
  bnStage(a2raw, e2g, e2bb, a2f16, 32, 32);
  stn_conv_wmma_t<27, 1, 4><<<cgrid, dim3(32), 0, stream>>>(a2f16, pe3, e3b, a3raw);
  bnStage(a3raw, e3g, e3bb, a3f16, 64, 64);
  stn_conv_wmma_t<1, 2, 2><<<cgrid, dim3(32), 0, stream>>>(a3f16, pc1, c1b, c1raw);
  bnStage(c1raw, c1g, c1bb, c1f16, 32, 32);
  stn_conv_wmma_t<1, 1, 2><<<cgrid, dim3(32), 0, stream>>>(c1f16, pc2, c2b, c2raw);
  bnStage(c2raw, c2g, c2bb, c2f16, 32, 32);

  stn_pool<<<dim3(CDIV(BT * 2048, 256)), dim3(256), 0, stream>>>(c2f16, feats, featsh);

  // LSTM: input projection for all 80 rows at once, then 5 recurrent steps
  stn_gemm_wmma_t<4><<<dim3(5, 8), dim3(32), 0, stream>>>(featsh, pl0i, gi, 80, 512, 2048, 2048, 512);

  hipMemsetAsync(h1st, 0, 16 * 128 * sizeof(_Float16), stream);
  hipMemsetAsync(h2st, 0, 16 * 128 * sizeof(_Float16), stream);

  for (int t = 0; t < TSTEPS; ++t) {
    stn_gemm_wmma_t<4><<<dim3(1, 8), dim3(32), 0, stream>>>(h1st, pl0h, gh1, 16, 512, 128, 128, 512);
    stn_lstm_gate<<<dim3(CDIV(2048, 256)), dim3(256), 0, stream>>>(gi + t * 512, TSTEPS * 512, gh1, l0_bih, l0_bhh,
                                                                   c1st, h1st, (_Float16*)nullptr, t);
    stn_gemm_wmma_t<4><<<dim3(1, 8), dim3(32), 0, stream>>>(h1st, pl1i, gi2, 16, 512, 128, 128, 512);
    stn_gemm_wmma_t<4><<<dim3(1, 8), dim3(32), 0, stream>>>(h2st, pl1h, gh2, 16, 512, 128, 128, 512);
    stn_lstm_gate<<<dim3(CDIV(2048, 256)), dim3(256), 0, stream>>>(gi2, 512, gh2, l1_bih, l1_bhh,
                                                                   c2st, h2st, h2seq, t);
  }

  // FC
  stn_gemm_wmma_t<4><<<dim3(5, 32), dim3(32), 0, stream>>>(h2seq, pfc, yb, 80, 2048, 128, 128, 2048);
  stn_add_bias<<<dim3(CDIV(BT * 2048, 256)), dim3(256), 0, stream>>>(yb, fcb);

  // decoder (yb viewed as (80, 32, 8, 4, 2))
  {
    size_t tot = (size_t)BT * 32 * 512;
    stn_convT<<<dim3((unsigned)CDIV(tot, 256)), dim3(256), 0, stream>>>(yb, d1w, d1b, d1buf, BT, 32, 32, 8, 4, 2, 0);
  }
  stn_bn_stats<<<dim3(32), dim3(256), 0, stream>>>(d1buf, stats, 32, 512, BT);
  {
    size_t tot = (size_t)BT * 32 * 512;
    stn_bn_apply_f32<<<dim3((unsigned)CDIV(tot, 256)), dim3(256), 0, stream>>>(d1buf, stats, d1g, d1bb, 32, 512, BT);
  }
  {
    size_t tot = (size_t)BT * 16 * 4096;
    stn_convT<<<dim3((unsigned)CDIV(tot, 256)), dim3(256), 0, stream>>>(d1buf, d2w, d2b, d2buf, BT, 32, 16, 16, 8, 4, 0);
  }
  stn_bn_stats<<<dim3(16), dim3(256), 0, stream>>>(d2buf, stats, 16, 4096, BT);
  {
    size_t tot = (size_t)BT * 16 * 4096;
    stn_bn_apply_f32<<<dim3((unsigned)CDIV(tot, 256)), dim3(256), 0, stream>>>(d2buf, stats, d2g, d2bb, 16, 4096, BT);
  }
  {
    size_t tot = (size_t)BT * 2 * 32768;
    stn_convT<<<dim3((unsigned)CDIV(tot, 256)), dim3(256), 0, stream>>>(d2buf, d3w, d3b, d3buf, BT, 16, 2, 32, 16, 8, 1);
  }

  // interp -> sort -> NMS -> masked output
  stn_interp<<<dim3(CDIV(48 * 2 * NVOX, 256)), dim3(256), 0, stream>>>(d3buf, frames);
  stn_sort<<<dim3(48), dim3(1024), 0, stream>>>(frames, orderG);
  stn_nms<<<dim3(48), dim3(32), 0, stream>>>(orderG, maskG);
  stn_final<<<dim3(CDIV(48 * 2 * NVOX, 256)), dim3(256), 0, stream>>>(frames, maskG, (float*)d_out);
}
